// FKR_EigenPro_50242527428615
// MI455X (gfx1250) — compile-verified, matrix operand order checked by reading, not run on hardware
//
#include <hip/hip_runtime.h>
#include <math.h>

// ---------------- problem / layout constants ----------------
#define DIM    128          // feature dim
#define NX     8192         // x rows
#define NCEN   50000        // real centers
#define NPAD   50176        // centers padded to multiple of 64
#define NWJ    10           // weight cols
#define NC     64           // centers staged per LDS chunk
#define NCH    (NPAD / NC)  // 784 chunks
#define LROW   136          // LDS row pitch in bf16 (272B) -> conflict-free ds_load_b128
#define MTILE  128          // x rows per workgroup (8 waves x 16)

// exp(-g*d2) = exp2(d2 * KSCALE),  g = 1/(2*5^2) = 0.02
#define KSCALE (-0.02f * 1.4426950408889634f)

typedef __bf16   bf16;
typedef float    v8f  __attribute__((ext_vector_type(8)));
typedef float    fv4  __attribute__((ext_vector_type(4)));
typedef bf16     v16b __attribute__((ext_vector_type(16)));
typedef bf16     bv4  __attribute__((ext_vector_type(4)));
typedef unsigned uv4  __attribute__((ext_vector_type(4)));
typedef int      iv4  __attribute__((ext_vector_type(4)));

// address-space-qualified pointer element types for the async-LDS builtin
typedef __attribute__((address_space(1))) iv4 g_iv4;   // global int4
typedef __attribute__((address_space(3))) iv4 l_iv4;   // LDS int4

union V16B { v16b v; uv4 q[2]; };

// ---------------- workspace layout (bytes) ----------------
#define XB_OFF  0u
#define XB_SZ   (NX * DIM * 2u)          // x as bf16
#define CB_OFF  (XB_OFF + XB_SZ)
#define CB_SZ   (NPAD * DIM * 2u)        // centers as bf16 (padded, zeroed tail)
#define WT_OFF  (CB_OFF + CB_SZ)
#define WT_SZ   (16u * NPAD * 2u)        // W^T padded to [16][NPAD] bf16
#define C2_OFF  (WT_OFF + WT_SZ)
#define C2_SZ   (NPAD * 4u)              // ||c||^2 (padded, zero tail)
#define X2_OFF  (C2_OFF + C2_SZ)
#define X2_SZ   (NX * 4u)                // ||x||^2

// ---------------- feature detection ----------------
#if defined(__has_builtin)
#if __has_builtin(__builtin_amdgcn_global_load_async_to_lds_b128)
#define HAVE_ASYNC_LDS 1
#endif
#endif

#if defined(__has_builtin) && __has_builtin(__builtin_amdgcn_exp2f)
#define EXP2F(x) __builtin_amdgcn_exp2f(x)
#else
#define EXP2F(x) exp2f(x)
#endif

__device__ __forceinline__ void wait_async_le4() {
#if defined(__has_builtin) && __has_builtin(__builtin_amdgcn_s_wait_asynccnt)
  __builtin_amdgcn_s_wait_asynccnt(4);
#else
  asm volatile("s_wait_asynccnt 4" ::: "memory");
#endif
}
__device__ __forceinline__ void wait_async_le0() {
#if defined(__has_builtin) && __has_builtin(__builtin_amdgcn_s_wait_asynccnt)
  __builtin_amdgcn_s_wait_asynccnt(0);
#else
  asm volatile("s_wait_asynccnt 0" ::: "memory");
#endif
}

__device__ __forceinline__ v8f wmma_bf16(v16b a, v16b b, v8f c) {
  // (neg_a, A, neg_b, B, c_mod, C, reuse_a, reuse_b)
  return __builtin_amdgcn_wmma_f32_16x16x32_bf16(false, a, false, b, (short)0, c,
                                                 false, false);
}

// ---------------- prep kernels ----------------
// one wave per row: f32 -> bf16 + row sum-of-squares
__global__ __launch_bounds__(256) void prep_x(const float* __restrict__ x,
                                              bf16* __restrict__ xb,
                                              float* __restrict__ x2) {
  const int lane = threadIdx.x & 31, w = threadIdx.x >> 5;
  const int row = blockIdx.x * 8 + w;
  fv4 v = *(const fv4*)(x + (size_t)row * DIM + lane * 4);
  *(bv4*)(xb + (size_t)row * DIM + lane * 4) = __builtin_convertvector(v, bv4);
  float ss = v[0] * v[0] + v[1] * v[1] + v[2] * v[2] + v[3] * v[3];
  for (int off = 16; off > 0; off >>= 1) ss += __shfl_xor(ss, off, 32);
  if (lane == 0) x2[row] = ss;
}

__global__ __launch_bounds__(256) void prep_c(const float* __restrict__ c,
                                              bf16* __restrict__ cb,
                                              float* __restrict__ c2) {
  const int lane = threadIdx.x & 31, w = threadIdx.x >> 5;
  const int row = blockIdx.x * 8 + w;
  if (row < NCEN) {
    fv4 v = *(const fv4*)(c + (size_t)row * DIM + lane * 4);
    *(bv4*)(cb + (size_t)row * DIM + lane * 4) = __builtin_convertvector(v, bv4);
    float ss = v[0] * v[0] + v[1] * v[1] + v[2] * v[2] + v[3] * v[3];
    for (int off = 16; off > 0; off >>= 1) ss += __shfl_xor(ss, off, 32);
    if (lane == 0) c2[row] = ss;
  } else {  // zero pad rows so padded columns contribute exactly 0
    bv4 z = {};
    *(bv4*)(cb + (size_t)row * DIM + lane * 4) = z;
    if (lane == 0) c2[row] = 0.f;
  }
}

// W^T padded: wt[j][n] = W[n][j] for j<10, n<NCEN else 0
__global__ __launch_bounds__(256) void prep_wt(const float* __restrict__ wgt,
                                               bf16* __restrict__ wt) {
  const int idx = blockIdx.x * 256 + threadIdx.x;   // 16*NPAD total
  const int j = idx / NPAD;
  const int n = idx - j * NPAD;
  float v = (j < NWJ && n < NCEN) ? wgt[(size_t)n * NWJ + j] : 0.f;
  wt[idx] = (bf16)v;
}

// ---------------- staging: centers chunk -> LDS ----------------
__device__ __forceinline__ void stage_chunk(const bf16* __restrict__ cb,
                                            bf16* smem, int t, int bufi, int c0) {
  bf16* base = smem + bufi * (NC * LROW);
#pragma unroll
  for (int i = 0; i < 4; ++i) {       // 64 rows * 16 chunks(16B) / 256 thr = 4
    const int idx = i * 256 + t;
    const int r = idx >> 4;
    const int kc = idx & 15;
    const bf16* g = cb + (size_t)(c0 + r) * DIM + kc * 8;
    bf16* l = base + r * LROW + kc * 8;
#ifdef HAVE_ASYNC_LDS
    __builtin_amdgcn_global_load_async_to_lds_b128((g_iv4*)g, (l_iv4*)l, 0, 0);
#else
    *(uv4*)l = *(const uv4*)g;
#endif
  }
}

// ---------------- main fused kernel ----------------
// Per wave: 16 x-rows resident as WMMA B operand. Loop over center chunks:
//   Kt(2 tiles of 16 centers) = centers x x   (8x v_wmma bf16, f32 acc)
//   d2 = x2 + c2 - 2*dot ; k = exp2(KSCALE*d2) ; pack -> B operand (no shuffles)
//   P^T += W^T x Kt                            (1x v_wmma bf16)
__global__ __launch_bounds__(256, 1) void fkr_main(
    const bf16* __restrict__ xb, const bf16* __restrict__ cb,
    const bf16* __restrict__ wt, const float* __restrict__ c2,
    const float* __restrict__ x2, float* __restrict__ out) {
  __shared__ __align__(16) bf16 smem[2 * NC * LROW];

  const int t = threadIdx.x;
  const int lane = t & 31;
  const int w = t >> 5;
  const int n16 = lane & 15;
  const int koff = (lane >> 4) * 8;          // bf16-element k offset for this half
  const int xrow = blockIdx.x * MTILE + w * 16 + n16;

  // resident B operand (x rows), 4 k-steps of 32
  V16B xv[4];
#pragma unroll
  for (int ks = 0; ks < 4; ++ks) {
    const bf16* p = xb + (size_t)xrow * DIM + ks * 32 + koff;
    xv[ks].q[0] = *(const uv4*)(p);
    xv[ks].q[1] = *(const uv4*)(p + 16);
  }
  const float x2l = x2[xrow];
  v8f pacc = {0.f, 0.f, 0.f, 0.f, 0.f, 0.f, 0.f, 0.f};

  stage_chunk(cb, smem, t, 0, 0);
  for (int c = 0; c < NCH; ++c) {
    const int cur = c & 1;
    if (c + 1 < NCH) stage_chunk(cb, smem, t, cur ^ 1, (c + 1) * NC);
#ifdef HAVE_ASYNC_LDS
    if (c + 1 < NCH) wait_async_le4();   // current chunk landed (in-order)
    else             wait_async_le0();
#endif
    __syncthreads();

    const bf16* buf = smem + cur * (NC * LROW);
    const int c0 = c * NC;
#pragma unroll
    for (int pr = 0; pr < NC / 32; ++pr) {
      const int cl = pr * 32;
      const int cg = c0 + cl;
      v8f a0 = {0.f, 0.f, 0.f, 0.f, 0.f, 0.f, 0.f, 0.f};
      v8f a1 = a0;
#pragma unroll
      for (int ks = 0; ks < 4; ++ks) {
        V16B A0, A1;
        const bf16* p0 = buf + (cl + n16) * LROW + ks * 32 + koff;
        A0.q[0] = *(const uv4*)(p0);
        A0.q[1] = *(const uv4*)(p0 + 16);
        const bf16* p1 = p0 + 16 * LROW;
        A1.q[0] = *(const uv4*)(p1);
        A1.q[1] = *(const uv4*)(p1 + 16);
        a0 = wmma_bf16(A0.v, xv[ks].v, a0);   // Kt tile0: centers cg..cg+15
        a1 = wmma_bf16(A1.v, xv[ks].v, a1);   // Kt tile1: centers cg+16..cg+31
      }
      // ||c||^2 for the 8 centers each half-lane owns, both tiles
      const float* c2p = c2 + cg + koff;
      fv4 ca0 = *(const fv4*)(c2p);
      fv4 ca1 = *(const fv4*)(c2p + 4);
      fv4 cb0 = *(const fv4*)(c2p + 16);
      fv4 cb1 = *(const fv4*)(c2p + 20);

      V16B kb;   // exp tile packed directly into 2nd-WMMA B-operand layout
#pragma unroll
      for (int v = 0; v < 8; ++v) {
        const float c2a = (v < 4) ? ca0[v] : ca1[v - 4];
        const float c2b = (v < 4) ? cb0[v] : cb1[v - 4];
        const float d0 = fmaxf(x2l + c2a - 2.f * a0[v], 0.f);
        const float d1 = fmaxf(x2l + c2b - 2.f * a1[v], 0.f);
        kb.v[v]     = (bf16)EXP2F(d0 * KSCALE);
        kb.v[v + 8] = (bf16)EXP2F(d1 * KSCALE);
      }
      // A operand: W^T tile [16 x 32], lane = output column j
      V16B wa;
      const bf16* wp = wt + (size_t)n16 * NPAD + cg + koff;
      wa.q[0] = *(const uv4*)(wp);
      wa.q[1] = *(const uv4*)(wp + 16);
      pacc = wmma_bf16(wa.v, kb.v, pacc);     // P^T accumulate
    }
    __syncthreads();
  }

  // P^T C-layout: lane<16 holds j=0..7 for its row; lane>=16 holds j=8..9
  float* op = out + (size_t)xrow * NWJ;
  if (lane < 16) {
#pragma unroll
    for (int v = 0; v < 8; ++v) op[v] = pacc[v];
  } else {
    op[8] = pacc[0];
    op[9] = pacc[1];
  }
}

// ---------------- host launcher ----------------
extern "C" void kernel_launch(void* const* d_in, const int* in_sizes, int n_in,
                              void* d_out, int out_size, void* d_ws, size_t ws_size,
                              hipStream_t stream) {
  const float* x = (const float*)d_in[0];        // [8192,128]
  const float* centers = (const float*)d_in[1];  // [50000,128]
  const float* weight = (const float*)d_in[2];   // [50000,10]
  float* out = (float*)d_out;                    // [8192,10]

  char* ws = (char*)d_ws;
  bf16* xb = (bf16*)(ws + XB_OFF);
  bf16* cb = (bf16*)(ws + CB_OFF);
  bf16* wt = (bf16*)(ws + WT_OFF);
  float* c2 = (float*)(ws + C2_OFF);
  float* x2 = (float*)(ws + X2_OFF);

  prep_x<<<NX / 8, 256, 0, stream>>>(x, xb, x2);
  prep_c<<<NPAD / 8, 256, 0, stream>>>(centers, cb, c2);
  prep_wt<<<(16 * NPAD) / 256, 256, 0, stream>>>(weight, wt);
  fkr_main<<<NX / MTILE, 256, 0, stream>>>(xb, cb, wt, c2, x2, out);
}
